// BiPyramid_72541997629776
// MI455X (gfx1250) — compile-verified, matrix operand order checked by reading, not run on hardware
//
#include <hip/hip_runtime.h>
#include <hip/hip_bf16.h>
#include <math.h>

// ---------------------------------------------------------------------------
// BiPyramid graph U-Net for MI455X (gfx1250, wave32, WMMA).
//
// Heavy compute = adj @ features batched GEMMs. Strategy:
//  * adjacency pyramid normalized once per level into f16, pre-swizzled into
//    the ISA 16x32 f16 A-fragment layout (CDNA5 ISA 7.12.2)
//  * feature linears write f16 pre-swizzled 32x16 B-fragments
//  * aggregation kernel: v_wmma_f32_16x16x32_f16 with f32 accumulate;
//    each wave owns a 16x32 output tile (one A fragment feeds two WMMAs),
//    software-pipelined so next-K fragment loads overlap the current WMMAs,
//    plus distance-2 prefetch on the dominant A stream
//  * all small C x C linears / pool / unpool / tanh stay on VALU (negligible)
// ---------------------------------------------------------------------------

typedef __attribute__((ext_vector_type(16))) _Float16 v16h;
typedef __attribute__((ext_vector_type(8)))  float    v8f;

#define BB 4
#define NN0 2048
#define CC0 32
#define HH 3
#define LL 2
#define EPSN 1e-12f

// ---- fragment packing helpers (ISA 7.12.2, 16-bit 16x32 A layout) ----------
// element e of lane: lanes split in halves of 16; half h, e<8 -> K = 8h+e,
// e>=8 -> K = 16 + 8h + (e-8).  Inverse used below.
__device__ __forceinline__ size_t apack_off(int b, int Nn, int i, int j) {
  int it = i >> 4, m = i & 15;
  int kt = j >> 5, K = j & 31;
  int half = (K >> 3) & 1;
  int e = (K & 7) + ((K & 16) ? 8 : 0);
  int lane = (half << 4) | m;
  return ((((size_t)b * (Nn >> 4) + it) * (Nn >> 5) + kt) * 32 + lane) * 16 + e;
}
// B (32x16) fragment: mirrored, lane indexes the N column.
__device__ __forceinline__ size_t bpack_off(int b, int Nn, int Cc, int j, int o) {
  int kt = j >> 5, K = j & 31;
  int vt = o >> 4, n = o & 15;
  int half = (K >> 3) & 1;
  int e = (K & 7) + ((K & 16) ? 8 : 0);
  int lane = (half << 4) | n;
  return ((((size_t)b * (Nn >> 5) + kt) * (Cc >> 4) + vt) * 32 + lane) * 16 + e;
}

// ---- adjacency pyramid -----------------------------------------------------
__global__ void pool_adj_kernel(const float* __restrict__ ain,
                                float* __restrict__ aout,
                                int Nout, size_t total) {
  size_t idx = (size_t)blockIdx.x * blockDim.x + threadIdx.x;
  if (idx >= total) return;
  int j = (int)(idx % Nout);
  size_t r = idx / Nout;
  int i = (int)(r % Nout);
  int b = (int)(r / Nout);
  int Nin = Nout * 2;
  const float* p = ain + ((size_t)b * Nin + 2 * i) * Nin + 2 * j;
  aout[idx] = fmaxf(fmaxf(p[0], p[1]), fmaxf(p[Nin], p[Nin + 1]));
}

// row-l2-normalize (axis=2) and pack into WMMA A fragments (f16)
__global__ void rownorm_pack_kernel(const float* __restrict__ adj,
                                    _Float16* __restrict__ Apack, int Nn) {
  int i = blockIdx.x, b = blockIdx.y, t = threadIdx.x;
  const float* row = adj + ((size_t)b * Nn + i) * Nn;
  __shared__ float red[256];
  float s = 0.f;
  for (int j = t; j < Nn; j += 256) { float v = row[j]; s += v * v; }
  red[t] = s; __syncthreads();
  for (int k = 128; k; k >>= 1) { if (t < k) red[t] += red[t + k]; __syncthreads(); }
  float inv = 1.0f / fmaxf(sqrtf(red[0]), EPSN);
  for (int j = t; j < Nn; j += 256)
    Apack[apack_off(b, Nn, i, j)] = (_Float16)(row[j] * inv);
}

// col-l2-normalize (axis=1), transpose, pack into A fragments:
// out logical [i][j] = adj[j][i] / ||adj[:,i]||
__global__ void colnorm_packT_kernel(const float* __restrict__ adj,
                                     _Float16* __restrict__ Apack, int Nn) {
  int i = blockIdx.x, b = blockIdx.y, t = threadIdx.x;  // i = column of adj
  const float* base = adj + (size_t)b * Nn * Nn + i;
  __shared__ float red[256];
  float s = 0.f;
  for (int j = t; j < Nn; j += 256) { float v = base[(size_t)j * Nn]; s += v * v; }
  red[t] = s; __syncthreads();
  for (int k = 128; k; k >>= 1) { if (t < k) red[t] += red[t + k]; __syncthreads(); }
  float inv = 1.0f / fmaxf(sqrtf(red[0]), EPSN);
  for (int j = t; j < Nn; j += 256)
    Apack[apack_off(b, Nn, i, j)] = (_Float16)(base[(size_t)j * Nn] * inv);
}

// ---- small linears ---------------------------------------------------------
// y = x @ w + b, written as f16 into WMMA B-fragment layout
__global__ void lin_pack_kernel(const float* __restrict__ x,
                                const float* __restrict__ w,
                                const float* __restrict__ bias,
                                _Float16* __restrict__ Xpack,
                                int Nn, int Cin, int Cout, size_t total) {
  size_t idx = (size_t)blockIdx.x * blockDim.x + threadIdx.x;
  if (idx >= total) return;
  int o = (int)(idx % Cout);
  int n = (int)((idx / Cout) % Nn);
  int b = (int)(idx / ((size_t)Cout * Nn));
  const float* xp = x + ((size_t)b * Nn + n) * Cin;
  float s = bias[o];
  for (int k = 0; k < Cin; ++k) s += xp[k] * w[(size_t)k * Cout + o];
  Xpack[bpack_off(b, Nn, Cout, n, o)] = (_Float16)s;
}

// y = (res ? res : 0) + x @ w + b  (f32)
__global__ void lin_f32_kernel(const float* __restrict__ x,
                               const float* __restrict__ w,
                               const float* __restrict__ bias,
                               const float* __restrict__ res,
                               float* __restrict__ y,
                               int Nn, int Cin, int Cout, size_t total) {
  size_t idx = (size_t)blockIdx.x * blockDim.x + threadIdx.x;
  if (idx >= total) return;
  int o = (int)(idx % Cout);
  int n = (int)((idx / Cout) % Nn);
  int b = (int)(idx / ((size_t)Cout * Nn));
  const float* xp = x + ((size_t)b * Nn + n) * Cin;
  float s = bias[o];
  for (int k = 0; k < Cin; ++k) s += xp[k] * w[(size_t)k * Cout + o];
  if (res) s += res[idx];
  y[idx] = s;
}

// ---- WMMA aggregation: Y[b,i,v] = sum_j Apack[b,i,j] * Xpack[b,j,v] --------
// 4 waves / block. Each wave owns a 16x32 output tile: one A fragment is
// reused by two WMMAs (halves adjacency traffic). Software pipelined: the
// three fragment loads for step k+1 are issued before the WMMAs of step k.
__global__ void agg_wmma_kernel(const _Float16* __restrict__ Apack,
                                const _Float16* __restrict__ Xpack,
                                float* __restrict__ Y, int Nn, int Cc) {
  const int lane = threadIdx.x & 31;
  const int it  = blockIdx.y * 4 + (threadIdx.x >> 5);
  const int vt2 = blockIdx.x;                 // pair of 16-wide v tiles
  const int b   = blockIdx.z;
  const int NT16 = Nn >> 4, NT32 = Nn >> 5, CT16 = Cc >> 4;
  const _Float16* Ab = Apack + ((((size_t)b * NT16 + it) * NT32) * 32 + lane) * 16;
  const _Float16* Xb = Xpack + ((((size_t)b * NT32) * CT16 + vt2 * 2) * 32 + lane) * 16;
  const size_t xstep = (size_t)CT16 * 512;

  v8f acc0 = {}, acc1 = {};
  v16h a  = *(const v16h*)(Ab);
  v16h x0 = *(const v16h*)(Xb);
  v16h x1 = *(const v16h*)(Xb + 512);
  for (int kt = 1; kt < NT32; ++kt) {
    Ab += 512;
    Xb += xstep;
    v16h an  = *(const v16h*)(Ab);
    v16h xn0 = *(const v16h*)(Xb);
    v16h xn1 = *(const v16h*)(Xb + 512);
    __builtin_prefetch(Ab + 512, 0, 3);       // distance-2 on the A stream
    acc0 = __builtin_amdgcn_wmma_f32_16x16x32_f16(
        false, a, false, x0, (short)0, acc0, false, false);
    acc1 = __builtin_amdgcn_wmma_f32_16x16x32_f16(
        false, a, false, x1, (short)0, acc1, false, false);
    a = an; x0 = xn0; x1 = xn1;
  }
  acc0 = __builtin_amdgcn_wmma_f32_16x16x32_f16(
      false, a, false, x0, (short)0, acc0, false, false);
  acc1 = __builtin_amdgcn_wmma_f32_16x16x32_f16(
      false, a, false, x1, (short)0, acc1, false, false);

  // C/D layout: VGPR r -> row r (lanes 0-15) / row r+8 (lanes 16-31), col=lane%16
  const int col = lane & 15, half = lane >> 4;
  float* Yp = Y + (((size_t)b * Nn) + it * 16) * Cc + vt2 * 32 + col;
#pragma unroll
  for (int r = 0; r < 8; ++r) {
    Yp[(size_t)(half * 8 + r) * Cc]      = acc0[r];
    Yp[(size_t)(half * 8 + r) * Cc + 16] = acc1[r];
  }
}

// ---- fused concat + upd linear + tanh --------------------------------------
__global__ void upd_tanh_kernel(const float* __restrict__ srcAgg,
                                const float* __restrict__ selfRep,
                                const float* __restrict__ dstAgg,
                                const float* __restrict__ w,   // (3C, C)
                                const float* __restrict__ bias,
                                float* __restrict__ y,
                                int Nn, int Cc, size_t total) {
  size_t idx = (size_t)blockIdx.x * blockDim.x + threadIdx.x;
  if (idx >= total) return;
  int o = (int)(idx % Cc);
  size_t bn = idx / Cc;  // flattened (b,n)
  const float* s0 = srcAgg  + bn * Cc;
  const float* s1 = selfRep + bn * Cc;
  const float* s2 = dstAgg  + bn * Cc;
  float s = bias[o];
  for (int k = 0; k < Cc; ++k) s += s0[k] * w[(size_t)k * Cc + o];
  for (int k = 0; k < Cc; ++k) s += s1[k] * w[(size_t)(Cc + k) * Cc + o];
  for (int k = 0; k < Cc; ++k) s += s2[k] * w[(size_t)(2 * Cc + k) * Cc + o];
  y[idx] = tanhf(s);
}

// ---- pool conv + pairwise node max ----------------------------------------
__global__ void pool_nodes_kernel(const float* __restrict__ x,
                                  const float* __restrict__ w,
                                  const float* __restrict__ bias,
                                  float* __restrict__ y,
                                  int NnOut, int Cin, int Cout, size_t total) {
  size_t idx = (size_t)blockIdx.x * blockDim.x + threadIdx.x;
  if (idx >= total) return;
  int o = (int)(idx % Cout);
  int n2 = (int)((idx / Cout) % NnOut);
  int b = (int)(idx / ((size_t)Cout * NnOut));
  const float* x0 = x + ((size_t)b * (2 * NnOut) + 2 * n2) * Cin;
  const float* x1 = x0 + Cin;
  float s0 = bias[o], s1 = bias[o];
  for (int k = 0; k < Cin; ++k) {
    float wk = w[(size_t)k * Cout + o];
    s0 += x0[k] * wk;
    s1 += x1[k] * wk;
  }
  y[idx] = fmaxf(s0, s1);
}

// ---- unpool: y[b,2n+t,o] = sum_c x[b,n,c]*wk[c,o,t] + bk[o] ---------------
__global__ void unpool_kernel(const float* __restrict__ x,
                              const float* __restrict__ wk,
                              const float* __restrict__ bk,
                              float* __restrict__ y,
                              int NnIn, int Cin, int Cout, size_t total) {
  size_t idx = (size_t)blockIdx.x * blockDim.x + threadIdx.x;
  if (idx >= total) return;
  int o = (int)(idx % Cout);
  int n2 = (int)((idx / Cout) % (2 * NnIn));
  int b = (int)(idx / ((size_t)Cout * 2 * NnIn));
  int n = n2 >> 1, t2 = n2 & 1;
  const float* xp = x + ((size_t)b * NnIn + n) * Cin;
  float s = bk[o];
  for (int c = 0; c < Cin; ++c) s += xp[c] * wk[((size_t)c * Cout + o) * 2 + t2];
  y[idx] = s;
}

// ===========================================================================
extern "C" void kernel_launch(void* const* d_in, const int* in_sizes, int n_in,
                              void* d_out, int out_size, void* d_ws, size_t ws_size,
                              hipStream_t stream) {
  (void)in_sizes; (void)n_in; (void)out_size; (void)ws_size;

  const int Ns[HH + 1] = {2048, 1024, 512, 256};
  const int Cs[HH + 1] = {32, 64, 128, 256};

  // ---- parse flattened pytree (insertion order: nodes, adj, params{layers_1,
  // layers_2, links_12}; lin={w,b}; sage={src,dst,self,upd}; unpool={wk,bk}) --
  int p = 2;
  auto nxt = [&]() { return (const float*)d_in[p++]; };
  struct Sage { const float *sw,*sb,*dw,*db,*ew,*eb,*uw,*ub; };
  Sage enc[HH + 1][LL], dec[HH + 1][LL];
  const float *convW[HH], *convB[HH], *upW[HH], *upB[HH];
  const float *lkW[(HH + 1) * LL + HH], *lkB[(HH + 1) * LL + HH];

  for (int i = 0; i <= HH; ++i) {
    for (int l = 0; l < LL; ++l) {
      Sage& s = enc[i][l];
      s.sw=nxt(); s.sb=nxt(); s.dw=nxt(); s.db=nxt();
      s.ew=nxt(); s.eb=nxt(); s.uw=nxt(); s.ub=nxt();
    }
    if (i < HH) { convW[i]=nxt(); convB[i]=nxt(); }
  }
  for (int i = 0; i <= HH; ++i) {
    if (i > 0) { upW[i-1]=nxt(); upB[i-1]=nxt(); }
    for (int l = 0; l < LL; ++l) {
      Sage& s = dec[i][l];
      s.sw=nxt(); s.sb=nxt(); s.dw=nxt(); s.db=nxt();
      s.ew=nxt(); s.eb=nxt(); s.uw=nxt(); s.ub=nxt();
    }
  }
  { int q = 0;
    for (int i = 0; i <= HH; ++i) {
      if (i > 0) { lkW[q]=nxt(); lkB[q]=nxt(); ++q; }
      for (int l = 0; l < LL; ++l) { lkW[q]=nxt(); lkB[q]=nxt(); ++q; }
    } }

  // ---- workspace arena (~127 MB) ------------------------------------------
  char* ws = (char*)d_ws;
  size_t off = 0;
  auto alloc = [&](size_t bytes) -> void* {
    off = (off + 255) & ~(size_t)255;
    void* r = ws + off; off += bytes; return r;
  };
  const float* adjL[HH + 1];
  adjL[0] = (const float*)d_in[1];
  float* adjLm[HH + 1] = {nullptr, nullptr, nullptr, nullptr};
  for (int s = 1; s <= HH; ++s) {
    adjLm[s] = (float*)alloc((size_t)BB * Ns[s] * Ns[s] * sizeof(float));
    adjL[s] = adjLm[s];
  }
  _Float16 *Arow[HH + 1], *AcolT[HH + 1];
  for (int s = 0; s <= HH; ++s) {
    Arow[s]  = (_Float16*)alloc((size_t)BB * Ns[s] * Ns[s] * sizeof(_Float16));
    AcolT[s] = (_Float16*)alloc((size_t)BB * Ns[s] * Ns[s] * sizeof(_Float16));
  }
  const size_t nodeF = (size_t)BB * NN0 * CC0;              // 262144 floats/level
  float* stash[11];
  for (int k = 0; k < 11; ++k) stash[k] = (float*)alloc(nodeF * sizeof(float));
  _Float16* srcPack = (_Float16*)alloc(nodeF * sizeof(_Float16));
  _Float16* dstPack = (_Float16*)alloc(nodeF * sizeof(_Float16));
  float* selfRep = (float*)alloc(nodeF * sizeof(float));
  float* srcAgg  = (float*)alloc(nodeF * sizeof(float));
  float* dstAgg  = (float*)alloc(nodeF * sizeof(float));
  float* xlink   = (float*)alloc(nodeF * sizeof(float));
  float* bufA    = (float*)alloc(nodeF * sizeof(float));
  float* bufB    = (float*)alloc(nodeF * sizeof(float));

  // ---- adjacency pyramid + normalized, WMMA-packed f16 levels --------------
  for (int s = 1; s <= HH; ++s) {
    size_t tot = (size_t)BB * Ns[s] * Ns[s];
    pool_adj_kernel<<<dim3((tot + 255) / 256), 256, 0, stream>>>(
        adjL[s - 1], adjLm[s], Ns[s], tot);
  }
  for (int s = 0; s <= HH; ++s) {
    dim3 gn(Ns[s], BB);
    rownorm_pack_kernel<<<gn, 256, 0, stream>>>(adjL[s], Arow[s], Ns[s]);
    colnorm_packT_kernel<<<gn, 256, 0, stream>>>(adjL[s], AcolT[s], Ns[s]);
  }

  // ---- one SAGE layer ------------------------------------------------------
  auto run_sage = [&](const Sage& P, const float* xin, float* xout, int s) {
    int Nn = Ns[s], Cc = Cs[s];
    size_t tot = (size_t)BB * Nn * Cc;
    dim3 g1((tot + 255) / 256);
    lin_pack_kernel<<<g1, 256, 0, stream>>>(xin, P.sw, P.sb, srcPack, Nn, Cc, Cc, tot);
    lin_pack_kernel<<<g1, 256, 0, stream>>>(xin, P.dw, P.db, dstPack, Nn, Cc, Cc, tot);
    lin_f32_kernel<<<g1, 256, 0, stream>>>(xin, P.ew, P.eb, nullptr, selfRep, Nn, Cc, Cc, tot);
    dim3 ga(Cc / 32, (Nn / 16) / 4, BB);
    agg_wmma_kernel<<<ga, 128, 0, stream>>>(Arow[s],  srcPack, srcAgg, Nn, Cc);
    agg_wmma_kernel<<<ga, 128, 0, stream>>>(AcolT[s], dstPack, dstAgg, Nn, Cc);
    upd_tanh_kernel<<<g1, 256, 0, stream>>>(srcAgg, selfRep, dstAgg, P.uw, P.ub, xout, Nn, Cc, tot);
  };

  // ---- encoder -------------------------------------------------------------
  const float* x = (const float*)d_in[0];
  int k = 0;
  for (int i = 0; i <= HH; ++i) {
    for (int l = 0; l < LL; ++l) { run_sage(enc[i][l], x, stash[k], i); x = stash[k]; ++k; }
    if (i < HH) {
      size_t tot = (size_t)BB * Ns[i + 1] * Cs[i + 1];
      pool_nodes_kernel<<<dim3((tot + 255) / 256), 256, 0, stream>>>(
          x, convW[i], convB[i], stash[k], Ns[i + 1], Cs[i], Cs[i + 1], tot);
      x = stash[k]; ++k;
    }
  }

  // ---- decoder -------------------------------------------------------------
  const float* cur = stash[10];
  float* pong[2] = {bufA, bufB};
  int pb = 0, li = 0;
  for (int i = 0; i <= HH; ++i) {
    int s = HH - i;  // level of this stage's sage layers
    if (i > 0) {
      int NnIn = Ns[s + 1], Cin = Cs[s + 1], Cout = Cs[s];
      size_t tot = (size_t)BB * NnIn * Cin;
      lin_f32_kernel<<<dim3((tot + 255) / 256), 256, 0, stream>>>(
          stash[10 - li], lkW[li], lkB[li], cur, xlink, NnIn, Cin, Cin, tot);
      float* out = pong[pb]; pb ^= 1;
      size_t tot2 = (size_t)BB * NnIn * 2 * Cout;
      unpool_kernel<<<dim3((tot2 + 255) / 256), 256, 0, stream>>>(
          xlink, upW[i - 1], upB[i - 1], out, NnIn, Cin, Cout, tot2);
      cur = out; ++li;
    }
    for (int l = 0; l < LL; ++l) {
      int Nn = Ns[s], Cc = Cs[s];
      size_t tot = (size_t)BB * Nn * Cc;
      lin_f32_kernel<<<dim3((tot + 255) / 256), 256, 0, stream>>>(
          stash[10 - li], lkW[li], lkB[li], cur, xlink, Nn, Cc, Cc, tot);
      float* out = pong[pb]; pb ^= 1;
      run_sage(dec[i][l], xlink, out, s);
      cur = out; ++li;
    }
  }

  // ---- output: (final nodes, original adjacency) ---------------------------
  hipMemcpyAsync(d_out, cur, nodeF * sizeof(float), hipMemcpyDeviceToDevice, stream);
  hipMemcpyAsync((float*)d_out + nodeF, d_in[1],
                 (size_t)BB * NN0 * NN0 * sizeof(float),
                 hipMemcpyDeviceToDevice, stream);
}
